// DSTN_29162827940295
// MI455X (gfx1250) — matrix operand order repaired and checked by reading, not compiled, not run on hardware
//
#include <hip/hip_runtime.h>
#include <math.h>

#define NROWS 8192
#define DDIM  512
#define KCL   20
#define KSTAGE 128

typedef __attribute__((ext_vector_type(16))) __bf16 v16bf;
typedef __attribute__((ext_vector_type(8)))  float  v8f;
typedef unsigned int u32x4 __attribute__((ext_vector_type(4)));
typedef int          i32x8 __attribute__((ext_vector_type(8)));
typedef int          i32x4 __attribute__((ext_vector_type(4)));

union V16 { uint4 q[2]; v16bf v; };
union V8o { __bf16 h[8]; uint4 q; };

__device__ inline unsigned int pack2bf(float a, float b) {
  __bf16 x = (__bf16)a, y = (__bf16)b;
  unsigned short sx, sy;
  __builtin_memcpy(&sx, &x, 2);
  __builtin_memcpy(&sy, &y, 2);
  return (unsigned int)sx | ((unsigned int)sy << 16);
}

// ---------------- degree / d^{-1/2} : one wave per row -------------------
__global__ __launch_bounds__(256) void deg_kernel(const float* __restrict__ adj,
                                                  float* __restrict__ dinv) {
  int wid  = (blockIdx.x * blockDim.x + threadIdx.x) >> 5;
  int lane = threadIdx.x & 31;
  if (wid >= NROWS) return;
  const float* row = adj + (size_t)wid * NROWS;
  float s = 0.f;
  for (int j = lane; j < NROWS; j += 32) s += row[j];
  #pragma unroll
  for (int off = 16; off > 0; off >>= 1) s += __shfl_xor(s, off, 32);
  if (lane == 0) dinv[wid] = rsqrtf(s + 1.0f);  // +1 for A_hat = A + I
}

// ---------------- cluster squared norms ----------------------------------
__global__ void cnorm_kernel(const float* __restrict__ cl, float* __restrict__ cn) {
  int k = blockIdx.x, lane = threadIdx.x;
  float s = 0.f;
  for (int d = lane; d < DDIM; d += 32) { float v = cl[k * DDIM + d]; s += v * v; }
  #pragma unroll
  for (int off = 16; off > 0; off >>= 1) s += __shfl_xor(s, off, 32);
  if (lane == 0) cn[k] = s;
}

// ---------------- fp32 -> bf16 convert -----------------------------------
__global__ __launch_bounds__(256) void cvt_kernel(const float* __restrict__ src,
                                                  __bf16* __restrict__ dst, int n) {
  int i = blockIdx.x * blockDim.x + threadIdx.x;
  if (i < n) dst[i] = (__bf16)src[i];
}

// ---------------- w -> w^T (bf16) ----------------------------------------
__global__ __launch_bounds__(256) void wT_kernel(const float* __restrict__ w,
                                                 __bf16* __restrict__ wT) {
  int idx = blockIdx.x * blockDim.x + threadIdx.x;   // idx = n*512 + k
  if (idx >= DDIM * DDIM) return;
  int n = idx >> 9, k = idx & (DDIM - 1);
  wT[idx] = (__bf16)w[k * DDIM + n];
}

// ---------------- Anorm precompute: bf16 normalized adjacency ------------
// anorm[i][j] = dinv[i]*dinv[j]*(adj[i][j] + (i==j)); 4 elems/thread.
__global__ __launch_bounds__(256) void anorm_kernel(const float* __restrict__ adj,
                                                    const float* __restrict__ dinv,
                                                    __bf16* __restrict__ out) {
  size_t base = ((size_t)blockIdx.x * 256 + threadIdx.x) * 4;
  int i = (int)(base >> 13);
  int j = (int)(base & (NROWS - 1));
  float di = dinv[i];
  float4 a = *(const float4*)(adj + base);
  float v0 = (a.x + (i == j     ? 1.f : 0.f)) * di * dinv[j];
  float v1 = (a.y + (i == j + 1 ? 1.f : 0.f)) * di * dinv[j + 1];
  float v2 = (a.z + (i == j + 2 ? 1.f : 0.f)) * di * dinv[j + 2];
  float v3 = (a.w + (i == j + 3 ? 1.f : 0.f)) * di * dinv[j + 3];
  union { __bf16 h[4]; uint2 u; } o;
  o.h[0] = (__bf16)v0; o.h[1] = (__bf16)v1; o.h[2] = (__bf16)v2; o.h[3] = (__bf16)v3;
  *(uint2*)(out + base) = o.u;
}

// ---------------- GEMM1: uT = (h @ w)^T  (M=8192,N=512,K=512) ------------
// one wave per 16(M) x 64(N) strip; software-pipelined fragment loads.
__global__ __launch_bounds__(256) void gemm1_kernel(const __bf16* __restrict__ h,
                                                    const __bf16* __restrict__ wT,
                                                    __bf16* __restrict__ uT) {
  int wid  = blockIdx.x * 8 + (threadIdx.x >> 5);
  int lane = threadIdx.x & 31;
  int m0 = (wid >> 3) * 16;
  int n0 = (wid & 7) * 64;
  int hf  = lane >> 4;
  int l16 = lane & 15;
  v8f zero = {0.f,0.f,0.f,0.f,0.f,0.f,0.f,0.f};
  v8f acc[4];
  #pragma unroll
  for (int t = 0; t < 4; ++t) acc[t] = zero;

  const __bf16* arow = h + (size_t)(m0 + l16) * DDIM + hf * 8;
  auto ldA = [&](int k0, V16& a) {
    a.q[0] = *(const uint4*)(arow + k0);
    a.q[1] = *(const uint4*)(arow + k0 + 16);
  };
  auto ldB = [&](int k0, V16* b) {
    #pragma unroll
    for (int t = 0; t < 4; ++t) {
      const uint4* qb = (const uint4*)(wT + (size_t)(n0 + 16 * t + l16) * DDIM + k0 + hf * 8);
      b[t].q[0] = qb[0]; b[t].q[1] = qb[2];
    }
  };

  V16 a0, a1, b0[4], b1[4];
  ldA(0, a0); ldB(0, b0);
  for (int k0 = 0; k0 < DDIM; k0 += 64) {
    ldA(k0 + 32, a1); ldB(k0 + 32, b1);
    #pragma unroll
    for (int t = 0; t < 4; ++t)
      acc[t] = __builtin_amdgcn_wmma_f32_16x16x32_bf16(
          false, a0.v, false, b0[t].v, (short)0, acc[t], false, false);
    if (k0 + 64 < DDIM) { ldA(k0 + 64, a0); ldB(k0 + 64, b0); }
    #pragma unroll
    for (int t = 0; t < 4; ++t)
      acc[t] = __builtin_amdgcn_wmma_f32_16x16x32_bf16(
          false, a1.v, false, b1[t].v, (short)0, acc[t], false, false);
  }
  #pragma unroll
  for (int t = 0; t < 4; ++t) {
    V8o o;
    #pragma unroll
    for (int g = 0; g < 8; ++g) o.h[g] = (__bf16)acc[t][g];
    *(uint4*)(uT + (size_t)(n0 + 16 * t + l16) * NROWS + m0 + hf * 8) = o.q;
  }
}

// ---------------- GEMM2 (TDM path): h' = relu(Anorm @ u) -----------------
// A tiles (16 x 128 bf16) DMA'd into double-buffered LDS by the Tensor Data
// Mover (wave 0 issues descriptors; s_wait_tensorcnt + barriers sync).
__global__ __launch_bounds__(256) void gemm2_tdm_kernel(const __bf16* __restrict__ anorm,
                                                        const __bf16* __restrict__ uT,
                                                        __bf16* __restrict__ hout,
                                                        float* __restrict__ fout,
                                                        int writeF) {
  __shared__ __align__(16) __bf16 ldsA[2][16 * KSTAGE];

  int m0   = blockIdx.x * 16;
  int tid  = threadIdx.x;
  int wv   = tid >> 5;
  int lane = tid & 31;
  int hf   = lane >> 4;
  int l16  = lane & 15;
  int n0   = wv * 64;

  unsigned long long abase =
      (unsigned long long)(uintptr_t)anorm + (unsigned long long)m0 * NROWS * 2ull;
  unsigned lds0 = (unsigned)(uintptr_t)&ldsA[0][0];
  unsigned lds1 = (unsigned)(uintptr_t)&ldsA[1][0];

  // D# group1: mask=0, data_size=1(2B), dims 8192x8192, tile 128x16,
  // dim0 stride 8192 (fields packed per ISA 8.4)
  i32x8 g1 = {0x00010000, 0x20000000, 0x20000000, 0x00800000, 16, NROWS, 0, 0};
  i32x4 z4 = {0, 0, 0, 0};
  i32x8 z8 = {0, 0, 0, 0, 0, 0, 0, 0};
  auto issue = [&](int stage, unsigned ldsaddr) {
    unsigned long long ga = abase + (unsigned long long)stage * KSTAGE * 2ull;
    u32x4 g0 = {1u, ldsaddr, (unsigned)ga, (unsigned)(ga >> 32) | 0x80000000u}; // type=2
    __builtin_amdgcn_tensor_load_to_lds(g0, g1, z4, z4, z8, 0);
  };

  bool issuer = (wv == 0);   // wave-uniform
  if (issuer) {
    issue(0, lds0);
    issue(1, lds1);
    __builtin_amdgcn_s_wait_tensorcnt((short)1);
  }
  __syncthreads();

  v8f zero = {0.f,0.f,0.f,0.f,0.f,0.f,0.f,0.f};
  v8f acc[4];
  #pragma unroll
  for (int t = 0; t < 4; ++t) acc[t] = zero;

  auto ldB = [&](int k0, V16* b) {
    #pragma unroll
    for (int t = 0; t < 4; ++t) {
      const uint4* qb = (const uint4*)(uT + (size_t)(n0 + 16 * t + l16) * NROWS + k0 + hf * 8);
      b[t].q[0] = qb[0]; b[t].q[1] = qb[2];
    }
  };

  const int NST = NROWS / KSTAGE;   // 64
  for (int s = 0; s < NST; ++s) {
    const __bf16* abuf = ldsA[s & 1];
    int kbase = s * KSTAGE;
    V16 bbuf[2][4];
    ldB(kbase, bbuf[0]);
    #pragma unroll
    for (int ks = 0; ks < 4; ++ks) {
      if (ks < 3) ldB(kbase + (ks + 1) * 32, bbuf[(ks + 1) & 1]);
      V16 ua;
      const uint4* lp = (const uint4*)(abuf + l16 * KSTAGE + ks * 32 + hf * 8);
      ua.q[0] = lp[0];   // K chunk base..base+7
      ua.q[1] = lp[2];   // K chunk base+16..base+23
      #pragma unroll
      for (int t = 0; t < 4; ++t)
        acc[t] = __builtin_amdgcn_wmma_f32_16x16x32_bf16(
            false, ua.v, false, bbuf[ks & 1][t].v, (short)0, acc[t], false, false);
    }
    __syncthreads();                       // all waves done reading this buffer
    if (issuer) {
      if (s + 2 < NST) {
        issue(s + 2, (s & 1) ? lds1 : lds0);
        __builtin_amdgcn_s_wait_tensorcnt((short)1);   // next buffer landed
      } else if (s + 1 < NST) {
        __builtin_amdgcn_s_wait_tensorcnt((short)0);
      }
    }
    __syncthreads();
  }

  #pragma unroll
  for (int t = 0; t < 4; ++t) {
    int n = n0 + 16 * t + l16;
    #pragma unroll
    for (int g = 0; g < 8; ++g) {
      float v = acc[t][g];
      v = v > 0.f ? v : 0.f;
      int m = m0 + hf * 8 + g;
      hout[(size_t)m * DDIM + n] = (__bf16)v;
      if (writeF) fout[(size_t)m * DDIM + n] = v;
    }
  }
}

// ---------------- GEMM2 (fallback): on-the-fly scaled adj ----------------
__global__ __launch_bounds__(256) void gemm2_fly_kernel(const float* __restrict__ adj,
                                                        const float* __restrict__ dinv,
                                                        const __bf16* __restrict__ uT,
                                                        __bf16* __restrict__ hout,
                                                        float* __restrict__ fout,
                                                        int writeF) {
  __shared__ __align__(16) unsigned int ldsA[2][16 * 24];  // 16 rows x 48 bf16 (padded)

  int m0   = blockIdx.x * 16;
  int tid  = threadIdx.x;
  int wv   = tid >> 5;
  int lane = tid & 31;
  int hf   = lane >> 4;
  int l16  = lane & 15;
  int n0   = wv * 64;

  int e  = tid * 2;
  int lr = e >> 5;
  int lc = e & 31;
  int i  = m0 + lr;
  float di = dinv[i];
  const float* arow = adj + (size_t)i * NROWS;

  v8f zero = {0.f,0.f,0.f,0.f,0.f,0.f,0.f,0.f};
  v8f acc[4];
  #pragma unroll
  for (int t = 0; t < 4; ++t) acc[t] = zero;

  auto loadTile = [&](int k0, int buf) {
    int j  = k0 + lc;
    float a0 = arow[j];
    float a1 = arow[j + 1];
    if (i == j)     a0 += 1.0f;
    if (i == j + 1) a1 += 1.0f;
    a0 *= di * dinv[j];
    a1 *= di * dinv[j + 1];
    ldsA[buf][lr * 24 + (lc >> 1)] = pack2bf(a0, a1);
  };

  loadTile(0, 0);
  __syncthreads();

  const int KT = NROWS / 32;
  for (int kk = 0; kk < KT; ++kk) {
    int k0 = kk * 32;
    if (kk + 1 < KT) {
      loadTile(k0 + 32, (kk + 1) & 1);
      if (kk + 2 < KT) __builtin_prefetch(arow + k0 + 64, 0, 0);
    }
    int buf = kk & 1;
    const uint4* lp = (const uint4*)&ldsA[buf][l16 * 24 + hf * 4];
    V16 ua;
    ua.q[0] = lp[0];
    ua.q[1] = lp[2];
    #pragma unroll
    for (int t = 0; t < 4; ++t) {
      const __bf16* bp = uT + (size_t)(n0 + 16 * t + l16) * NROWS + k0 + hf * 8;
      V16 ub;
      const uint4* qb = (const uint4*)bp;
      ub.q[0] = qb[0]; ub.q[1] = qb[2];
      acc[t] = __builtin_amdgcn_wmma_f32_16x16x32_bf16(
          false, ua.v, false, ub.v, (short)0, acc[t], false, false);
    }
    __syncthreads();
  }

  #pragma unroll
  for (int t = 0; t < 4; ++t) {
    int n = n0 + 16 * t + l16;
    #pragma unroll
    for (int g = 0; g < 8; ++g) {
      float v = acc[t][g];
      v = v > 0.f ? v : 0.f;
      int m = m0 + hf * 8 + g;
      hout[(size_t)m * DDIM + n] = (__bf16)v;
      if (writeF) fout[(size_t)m * DDIM + n] = v;
    }
  }
}

// ---------------- student-t soft assignment: one wave per row ------------
__global__ __launch_bounds__(256) void q_kernel(const float* __restrict__ h,
                                                const float* __restrict__ cl,
                                                const float* __restrict__ cn,
                                                float* __restrict__ q) {
  int wid  = (blockIdx.x * blockDim.x + threadIdx.x) >> 5;
  int lane = threadIdx.x & 31;
  if (wid >= NROWS) return;
  const float* hr = h + (size_t)wid * DDIM;
  float acc[KCL];
  #pragma unroll
  for (int k = 0; k < KCL; ++k) acc[k] = 0.f;
  float hn = 0.f;
  for (int d = lane; d < DDIM; d += 32) {
    float hv = hr[d];
    hn += hv * hv;
    #pragma unroll
    for (int k = 0; k < KCL; ++k) acc[k] += hv * cl[k * DDIM + d];
  }
  #pragma unroll
  for (int off = 16; off > 0; off >>= 1) {
    hn += __shfl_xor(hn, off, 32);
    #pragma unroll
    for (int k = 0; k < KCL; ++k) acc[k] += __shfl_xor(acc[k], off, 32);
  }
  if (lane == 0) {
    float qs[KCL], s = 0.f;
    #pragma unroll
    for (int k = 0; k < KCL; ++k) {
      float sq = hn + cn[k] - 2.0f * acc[k];
      float t  = 1.0f / (1.0f + 0.5f * sq);   // v = 2
      float qq = t * sqrtf(t);                // t^{1.5}
      qs[k] = qq; s += qq;
    }
    float inv = 1.0f / s;
    #pragma unroll
    for (int k = 0; k < KCL; ++k) q[(size_t)wid * KCL + k] = qs[k] * inv;
  }
}

extern "C" void kernel_launch(void* const* d_in, const int* in_sizes, int n_in,
                              void* d_out, int out_size, void* d_ws, size_t ws_size,
                              hipStream_t stream) {
  const float* x       = (const float*)d_in[0];
  const float* adj     = (const float*)d_in[1];
  const float* w[4]    = {(const float*)d_in[2], (const float*)d_in[3],
                          (const float*)d_in[4], (const float*)d_in[5]};
  const float* cluster = (const float*)d_in[6];

  float* outH = (float*)d_out;
  float* outQ = outH + (size_t)NROWS * DDIM;

  char* base = (char*)d_ws;
  size_t off = 0;
  float*  dinv  = (float*)(base + off);  off += (size_t)NROWS * 4;
  float*  cn    = (float*)(base + off);  off += 256;
  __bf16* wT    = (__bf16*)(base + off); off += (size_t)DDIM * DDIM * 2;
  __bf16* hA    = (__bf16*)(base + off); off += (size_t)NROWS * DDIM * 2;
  __bf16* hB    = (__bf16*)(base + off); off += (size_t)NROWS * DDIM * 2;
  __bf16* uT    = (__bf16*)(base + off); off += (size_t)NROWS * DDIM * 2;
  __bf16* anorm = (__bf16*)(base + off);
  size_t needTDM = off + (size_t)NROWS * NROWS * 2;   // + 128 MB
  int useTDM = (ws_size >= needTDM) ? 1 : 0;
  (void)in_sizes; (void)n_in; (void)out_size;

  deg_kernel<<<NROWS / 8, 256, 0, stream>>>(adj, dinv);
  cnorm_kernel<<<KCL, 32, 0, stream>>>(cluster, cn);
  cvt_kernel<<<(NROWS * DDIM) / 256, 256, 0, stream>>>(x, hA, NROWS * DDIM);
  if (useTDM) {
    anorm_kernel<<<(int)(((size_t)NROWS * NROWS) / 1024), 256, 0, stream>>>(adj, dinv, anorm);
  }

  __bf16* hcur = hA;
  __bf16* hnxt = hB;
  for (int layer = 0; layer < 4; ++layer) {
    wT_kernel<<<(DDIM * DDIM) / 256, 256, 0, stream>>>(w[layer], wT);
    gemm1_kernel<<<NROWS / 16, 256, 0, stream>>>(hcur, wT, uT);
    if (useTDM) {
      gemm2_tdm_kernel<<<NROWS / 16, 256, 0, stream>>>(anorm, uT, hnxt, outH,
                                                       layer == 3 ? 1 : 0);
    } else {
      gemm2_fly_kernel<<<NROWS / 16, 256, 0, stream>>>(adj, dinv, uT, hnxt, outH,
                                                       layer == 3 ? 1 : 0);
    }
    __bf16* tmp = hcur; hcur = hnxt; hnxt = tmp;
  }

  q_kernel<<<NROWS / 8, 256, 0, stream>>>(outH, cluster, cn, outQ);
}
